// Attention_46746424050262
// MI455X (gfx1250) — matrix operand order repaired and checked by reading, not compiled, not run on hardware
//
#include <hip/hip_runtime.h>
#include <math.h>

#define DMODEL 1024
#define NHEADS 16
#define DHEAD  64
#define NTOK   2048
#define NBATCH 2
#define NROWS  (NBATCH * NTOK) /* 4096 */

typedef __attribute__((ext_vector_type(16))) __bf16 v16bf;
typedef __attribute__((ext_vector_type(8)))  __bf16 v8bf;
typedef __attribute__((ext_vector_type(8)))  float  v8f;

union V16U { v16bf v; v8bf h[2]; };

static __device__ __forceinline__ __bf16 f2bf(float f) {
  unsigned u = __builtin_bit_cast(unsigned, f);
  unsigned short h = (unsigned short)((u + 0x7FFFu + ((u >> 16) & 1u)) >> 16);
  return __builtin_bit_cast(__bf16, h);
}

static __device__ __forceinline__ v8bf pack8(const float4 a, const float4 b) {
  v8bf r;
  r[0] = f2bf(a.x); r[1] = f2bf(a.y); r[2] = f2bf(a.z); r[3] = f2bf(a.w);
  r[4] = f2bf(b.x); r[5] = f2bf(b.y); r[6] = f2bf(b.z); r[7] = f2bf(b.w);
  return r;
}

static __device__ __forceinline__ v8f zero8() {
  v8f z = {0.f, 0.f, 0.f, 0.f, 0.f, 0.f, 0.f, 0.f};
  return z;
}

static __device__ __forceinline__ v16bf ones16() {
  const __bf16 one = __builtin_bit_cast(__bf16, (unsigned short)0x3F80);
  v16bf r;
#pragma unroll
  for (int i = 0; i < 16; ++i) r[i] = one;
  return r;
}

static __device__ __forceinline__ v8f wmma_bf16(v16bf a, v16bf b, v8f c) {
  return __builtin_amdgcn_wmma_f32_16x16x32_bf16(false, a, false, b, (short)0, c,
                                                 false, false);
}

// Async copy of 32 bytes (global -> LDS), per lane: two b128 transfers.
// INST_OFFSET applies to both the LDS and global address (ISA 08 §4.4).
static __device__ __forceinline__ void async_copy32(unsigned lds_off,
                                                    unsigned long long gaddr) {
  asm volatile(
      "global_load_async_to_lds_b128 %0, %1, off\n\t"
      "global_load_async_to_lds_b128 %0, %1, off offset:16"
      :: "v"(lds_off), "v"(gaddr) : "memory");
}

// ---------------------------------------------------------------------------
// Prepass: fp32 -> bf16 (x and the four weight matrices), 8 elems/thread.
// ---------------------------------------------------------------------------
__global__ __launch_bounds__(256) void tobf16_kernel(
    const float* __restrict__ in, __bf16* __restrict__ out, int n8) {
  const int i = blockIdx.x * 256 + threadIdx.x;
  if (i < n8) {
    const float4* p = (const float4*)(in + (size_t)i * 8);
    *(v8bf*)(out + (size_t)i * 8) = pack8(p[0], p[1]);
  }
}

// ---------------------------------------------------------------------------
// Kernel 1: Y = xb @ Wb^T + bias (bf16 WMMA, f32 accum). W tile (128x32 bf16,
// 8 KB) double-buffered in LDS via GLOBAL_LOAD_ASYNC_TO_LDS_B128.
// MODE 0: Q -> [B,H,N,dh] * 1/sqrt(dh);  MODE 1: K -> [B,H,N,dh];
// MODE 2: V -> [B,H,dh,N] (transposed at store: packed 16B stores).
// ---------------------------------------------------------------------------
template <int MODE>
__global__ __launch_bounds__(256) void qkv_proj_kernel(
    const __bf16* __restrict__ xb, const __bf16* __restrict__ Wb,
    const float* __restrict__ bias, __bf16* __restrict__ out) {
  __shared__ __align__(64) __bf16 Wlds[2][128 * 32];
  const int tid = threadIdx.x;
  const int lane = tid & 31, lane16 = lane & 15, hi = lane >> 4;
  const int wid = tid >> 5;
  const int wave_m = wid >> 1, wave_n = wid & 1;
  const int gm0 = blockIdx.x * 64;
  const int wn0 = blockIdx.y * 128;

  // staging map: 256 threads x 32B = one 128x32 bf16 tile
  const int srow = tid >> 1, sh = tid & 1;
  const __bf16* wsrc = Wb + (size_t)(wn0 + srow) * DMODEL + sh * 16;
  const unsigned lds_b[2] = {
      (unsigned)(unsigned long long)(const void*)&Wlds[0][srow * 32 + sh * 16],
      (unsigned)(unsigned long long)(const void*)&Wlds[1][srow * 32 + sh * 16]};

  const __bf16* arow = xb + (size_t)(gm0 + wave_m * 16 + lane16) * DMODEL;

  async_copy32(lds_b[0], (unsigned long long)wsrc);  // prologue: tile 0

  v8f acc[4] = {zero8(), zero8(), zero8(), zero8()};
  for (int ks = 0; ks < DMODEL / 32; ++ks) {
    const int k = ks * 32;
    if (ks + 1 < DMODEL / 32) {
      async_copy32(lds_b[(ks + 1) & 1], (unsigned long long)(wsrc + k + 32));
      asm volatile("s_wait_asynccnt 2" ::: "memory");  // tile ks complete
    } else {
      asm volatile("s_wait_asynccnt 0" ::: "memory");
    }
    __syncthreads();

    V16U a;  // A frag: row = lane16, K-contig (k = 16*hi + s)
    a.h[0] = *(const v8bf*)(arow + k + hi * 16);
    a.h[1] = *(const v8bf*)(arow + k + 8 + hi * 16);
    const __bf16* wl = &Wlds[ks & 1][0];
#pragma unroll
    for (int nn = 0; nn < 4; ++nn) {
      const int wr = wave_n * 64 + nn * 16 + lane16;
      v16bf b = *(const v16bf*)&wl[wr * 32 + hi * 16];  // B frag: K-contig
      acc[nn] = wmma_bf16(a.v, b, acc[nn]);
    }
    __syncthreads();  // buffer reuse fence
  }

#pragma unroll
  for (int nn = 0; nn < 4; ++nn) {
    const int c = wn0 + wave_n * 64 + nn * 16 + lane16;  // output column
    const float bb = bias[c];
    const int h = c >> 6, dh = c & 63;
    const int r0 = gm0 + wave_m * 16 + hi * 8;  // first of 8 contiguous rows
    if (MODE == 2) {
      const int b = r0 >> 11, n = r0 & 2047;
      v8bf pk;
#pragma unroll
      for (int v = 0; v < 8; ++v) pk[v] = f2bf(acc[nn][v] + bb);
      *(v8bf*)(out + (((size_t)(b * NHEADS + h) * DHEAD + dh) * NTOK + n)) = pk;
    } else {
#pragma unroll
      for (int v = 0; v < 8; ++v) {
        const int r = r0 + v;
        const int b = r >> 11, n = r & 2047;
        float f = acc[nn][v] + bb;
        if (MODE == 0) f *= 0.125f;  // 1/sqrt(64) folded into Q
        out[(((size_t)(b * NHEADS + h) * NTOK + n) * DHEAD + dh)] = f2bf(f);
      }
    }
  }
}

// ---------------------------------------------------------------------------
// Kernel 2: causal flash attention. 4 waves/block, 32 queries/wave (two
// 16-row groups sharing K/V fragments), 32 keys/step. Tile-wise running max
// (exact for softmax) shared across all 32 rows; row sums via ones-WMMA.
// ---------------------------------------------------------------------------
template <bool MASKED>
static __device__ __forceinline__ void attn_step(
    int kb, int q0, int lane16, int hi, const __bf16* kbase, const __bf16* vbase,
    __bf16* pp /* per-wave [2][16*32] */, const V16U (&aq)[2][2],
    const v16bf ones, v8f (&acc)[2][4], float& mrun, float (&lrow)[2][8]) {
  // ---- K B-fragments (shared by both row groups)
  const __bf16* k0p = kbase + (size_t)kb * DHEAD;
  const __bf16* k1p = k0p + 16 * DHEAD;
  v16bf bk00 = *(const v16bf*)(k0p + hi * 16);       // keys kb+0..15, d 0..31
  v16bf bk01 = *(const v16bf*)(k0p + 32 + hi * 16);  // keys kb+0..15, d 32..63
  v16bf bk10 = *(const v16bf*)(k1p + hi * 16);       // keys kb+16..31
  v16bf bk11 = *(const v16bf*)(k1p + 32 + hi * 16);

  // ---- S = Q @ K^T : 2 row groups x 2 key tiles
  v8f s[2][2];
#pragma unroll
  for (int g = 0; g < 2; ++g) {
    v8f t0 = wmma_bf16(aq[g][0].v, bk00, zero8());
    s[g][0] = wmma_bf16(aq[g][1].v, bk01, t0);
    v8f t1 = wmma_bf16(aq[g][0].v, bk10, zero8());
    s[g][1] = wmma_bf16(aq[g][1].v, bk11, t1);
  }

  // ---- causal mask (diagonal block only); row = g*16 + v + 8*hi
  if (MASKED) {
#pragma unroll
    for (int g = 0; g < 2; ++g)
#pragma unroll
      for (int t = 0; t < 2; ++t)
#pragma unroll
        for (int v = 0; v < 8; ++v) {
          const int q = q0 + g * 16 + v + hi * 8;
          const int c = kb + t * 16 + lane16;
          s[g][t][v] = (c <= q) ? s[g][t][v] : -INFINITY;
        }
  }

  // ---- shared tile max over all 32 rows (exact softmax offset)
  float m = s[0][0][0];
#pragma unroll
  for (int g = 0; g < 2; ++g)
#pragma unroll
    for (int t = 0; t < 2; ++t)
#pragma unroll
      for (int v = 0; v < 8; ++v) m = fmaxf(m, s[g][t][v]);
  m = fmaxf(m, __shfl_xor(m, 1, 32));
  m = fmaxf(m, __shfl_xor(m, 2, 32));
  m = fmaxf(m, __shfl_xor(m, 4, 32));
  m = fmaxf(m, __shfl_xor(m, 8, 32));
  m = fmaxf(m, __shfl_xor(m, 16, 32));
  const float mn = fmaxf(mrun, m);
  const float scale = __expf(mrun - mn);  // first block: exp(-inf) == 0
  mrun = mn;

  // ---- rescale accumulators / sums by the scalar factor
#pragma unroll
  for (int g = 0; g < 2; ++g) {
#pragma unroll
    for (int nn = 0; nn < 4; ++nn)
#pragma unroll
      for (int v = 0; v < 8; ++v) acc[g][nn][v] *= scale;
#pragma unroll
    for (int v = 0; v < 8; ++v) lrow[g][v] *= scale;
  }

  // ---- P = exp(S - m): C-layout -> LDS -> A-layout (same-wave DS ordering)
#pragma unroll
  for (int g = 0; g < 2; ++g) {
    __bf16* pg = pp + g * 512;
#pragma unroll
    for (int v = 0; v < 8; ++v) {
      const int mr = v + hi * 8;
      pg[mr * 32 + lane16] = f2bf(__expf(s[g][0][v] - mn));
      pg[mr * 32 + 16 + lane16] = f2bf(__expf(s[g][1][v] - mn));
    }
  }
  asm volatile("s_wait_dscnt 0" ::: "memory");
  V16U ap[2];
#pragma unroll
  for (int g = 0; g < 2; ++g) {
    const __bf16* pg = pp + g * 512;
    ap[g].h[0] = *(const v8bf*)&pg[lane16 * 32 + hi * 16];
    ap[g].h[1] = *(const v8bf*)&pg[lane16 * 32 + 8 + hi * 16];
  }

  // ---- row sums on the tensor core: P @ ones -> C-layout row sums
#pragma unroll
  for (int g = 0; g < 2; ++g) {
    v8f rs = wmma_bf16(ap[g].v, ones, zero8());
#pragma unroll
    for (int v = 0; v < 8; ++v) lrow[g][v] += rs[v];
  }

  // ---- ctx += P @ V : V fragments shared by both row groups
  const __bf16* vt = vbase + kb;
#pragma unroll
  for (int nn = 0; nn < 4; ++nn) {
    v16bf bv = *(const v16bf*)(vt + (size_t)nn * 16 * NTOK);
#pragma unroll
    for (int g = 0; g < 2; ++g) acc[g][nn] = wmma_bf16(ap[g].v, bv, acc[g][nn]);
  }
}

__global__ __launch_bounds__(128) void attn_kernel(
    const __bf16* __restrict__ Q, const __bf16* __restrict__ K,
    const __bf16* __restrict__ Vt, __bf16* __restrict__ ctx) {
  __shared__ __align__(64) __bf16 Plds[4][2 * 16 * 32];
  const int tid = threadIdx.x;
  const int lane = tid & 31, lane16 = lane & 15, hi = lane >> 4;
  const int wid = tid >> 5;
  const int bh = blockIdx.y;                   // b*16 + h
  const int q0 = blockIdx.x * 128 + wid * 32;  // 32 queries per wave
  __bf16* pp = &Plds[wid][0];

  // Q A-fragments for both row groups (loop invariant)
  V16U aq[2][2];
#pragma unroll
  for (int g = 0; g < 2; ++g) {
    const __bf16* qrow = Q + ((size_t)bh * NTOK + q0 + g * 16 + lane16) * DHEAD;
    aq[g][0].h[0] = *(const v8bf*)(qrow + hi * 16);
    aq[g][0].h[1] = *(const v8bf*)(qrow + 8 + hi * 16);
    aq[g][1].h[0] = *(const v8bf*)(qrow + 32 + hi * 16);
    aq[g][1].h[1] = *(const v8bf*)(qrow + 40 + hi * 16);
  }

  const __bf16* kbase = K + ((size_t)bh * NTOK + lane16) * DHEAD;
  const __bf16* vbase = Vt + ((size_t)bh * DHEAD + lane16) * NTOK + hi * 16;
  const v16bf ones = ones16();

  v8f acc[2][4] = {{zero8(), zero8(), zero8(), zero8()},
                   {zero8(), zero8(), zero8(), zero8()}};
  float mrun = -INFINITY;
  float lrow[2][8];
#pragma unroll
  for (int g = 0; g < 2; ++g)
#pragma unroll
    for (int v = 0; v < 8; ++v) lrow[g][v] = 0.f;

  // q0 is a multiple of 32: blocks kb < q0 are fully unmasked, kb == q0 is
  // the (single) diagonal block.
  int kb = 0;
  for (; kb < q0; kb += 32)
    attn_step<false>(kb, q0, lane16, hi, kbase, vbase, pp, aq, ones, acc, mrun,
                     lrow);
  attn_step<true>(q0, q0, lane16, hi, kbase, vbase, pp, aq, ones, acc, mrun,
                  lrow);

  // ---- normalize and write ctx [B, N, DMODEL] as bf16
  const int b = bh >> 4, h = bh & 15;
#pragma unroll
  for (int g = 0; g < 2; ++g)
#pragma unroll
    for (int v = 0; v < 8; ++v) {
      const int q = q0 + g * 16 + v + hi * 8;
      const float rinv = 1.f / lrow[g][v];
      __bf16* orow = ctx + ((size_t)b * NTOK + q) * DMODEL + h * DHEAD + lane16;
#pragma unroll
      for (int nn = 0; nn < 4; ++nn) orow[nn * 16] = f2bf(acc[g][nn][v] * rinv);
    }
}

// ---------------------------------------------------------------------------
// Kernel 3a: out = ctx @ Wo^T + bo + x (f32 into d_out). Same async-staged
// tiling; ctx A-fragments load straight from global (already bf16).
// ---------------------------------------------------------------------------
__global__ __launch_bounds__(256) void oproj_kernel(
    const __bf16* __restrict__ ctx, const __bf16* __restrict__ Wob,
    const float* __restrict__ bo, const float* __restrict__ x,
    float* __restrict__ out) {
  __shared__ __align__(64) __bf16 Wlds[2][128 * 32];
  const int tid = threadIdx.x;
  const int lane = tid & 31, lane16 = lane & 15, hi = lane >> 4;
  const int wid = tid >> 5;
  const int wave_m = wid >> 1, wave_n = wid & 1;
  const int gm0 = blockIdx.x * 64;
  const int wn0 = blockIdx.y * 128;

  const int srow = tid >> 1, sh = tid & 1;
  const __bf16* wsrc = Wob + (size_t)(wn0 + srow) * DMODEL + sh * 16;
  const unsigned lds_b[2] = {
      (unsigned)(unsigned long long)(const void*)&Wlds[0][srow * 32 + sh * 16],
      (unsigned)(unsigned long long)(const void*)&Wlds[1][srow * 32 + sh * 16]};

  const __bf16* arow = ctx + (size_t)(gm0 + wave_m * 16 + lane16) * DMODEL;

  async_copy32(lds_b[0], (unsigned long long)wsrc);

  v8f acc[4] = {zero8(), zero8(), zero8(), zero8()};
  for (int ks = 0; ks < DMODEL / 32; ++ks) {
    const int k = ks * 32;
    if (ks + 1 < DMODEL / 32) {
      async_copy32(lds_b[(ks + 1) & 1], (unsigned long long)(wsrc + k + 32));
      asm volatile("s_wait_asynccnt 2" ::: "memory");
    } else {
      asm volatile("s_wait_asynccnt 0" ::: "memory");
    }
    __syncthreads();

    V16U a;
    a.h[0] = *(const v8bf*)(arow + k + hi * 16);
    a.h[1] = *(const v8bf*)(arow + k + 8 + hi * 16);
    const __bf16* wl = &Wlds[ks & 1][0];
#pragma unroll
    for (int nn = 0; nn < 4; ++nn) {
      const int wr = wave_n * 64 + nn * 16 + lane16;
      v16bf b = *(const v16bf*)&wl[wr * 32 + hi * 16];
      acc[nn] = wmma_bf16(a.v, b, acc[nn]);
    }
    __syncthreads();
  }
#pragma unroll
  for (int nn = 0; nn < 4; ++nn) {
    const int c = wn0 + wave_n * 64 + nn * 16 + lane16;
    const float bb = bo[c];
#pragma unroll
    for (int v = 0; v < 8; ++v) {
      const size_t r = (size_t)(gm0 + wave_m * 16 + v + hi * 8);
      out[r * DMODEL + c] = acc[nn][v] + bb + x[r * DMODEL + c];
    }
  }
}

// ---------------------------------------------------------------------------
// Kernel 3b: in-place LayerNorm over last dim (one row per 256-thread block).
// ---------------------------------------------------------------------------
__global__ __launch_bounds__(256) void ln_kernel(float* __restrict__ y,
                                                 const float* __restrict__ gamma,
                                                 const float* __restrict__ beta) {
  __shared__ float red[16];
  const int row = blockIdx.x, tid = threadIdx.x;
  float* yr = y + (size_t)row * DMODEL;
  float vals[4];
  float s = 0.f, sq = 0.f;
#pragma unroll
  for (int i = 0; i < 4; ++i) {
    vals[i] = yr[tid + i * 256];
    s += vals[i];
    sq += vals[i] * vals[i];
  }
#pragma unroll
  for (int m = 1; m < 32; m <<= 1) {
    s += __shfl_xor(s, m, 32);
    sq += __shfl_xor(sq, m, 32);
  }
  const int wid = tid >> 5;
  if ((tid & 31) == 0) { red[wid] = s; red[8 + wid] = sq; }
  __syncthreads();
  s = 0.f; sq = 0.f;
#pragma unroll
  for (int i = 0; i < 8; ++i) { s += red[i]; sq += red[8 + i]; }
  const float mean = s * (1.f / DMODEL);
  const float var = sq * (1.f / DMODEL) - mean * mean;
  const float rstd = rsqrtf(var + 1e-5f);
#pragma unroll
  for (int i = 0; i < 4; ++i) {
    const int c = tid + i * 256;
    yr[c] = (vals[i] - mean) * rstd * gamma[c] + beta[c];
  }
}

// ---------------------------------------------------------------------------
extern "C" void kernel_launch(void* const* d_in, const int* in_sizes, int n_in,
                              void* d_out, int out_size, void* d_ws,
                              size_t ws_size, hipStream_t stream) {
  (void)in_sizes; (void)n_in; (void)out_size; (void)ws_size;
  const float* x  = (const float*)d_in[0];
  const float* Wq = (const float*)d_in[1];
  const float* bq = (const float*)d_in[2];
  const float* Wk = (const float*)d_in[3];
  const float* bk = (const float*)d_in[4];
  const float* Wv = (const float*)d_in[5];
  const float* bv = (const float*)d_in[6];
  const float* Wo = (const float*)d_in[7];
  const float* bo = (const float*)d_in[8];
  const float* g  = (const float*)d_in[9];
  const float* be = (const float*)d_in[10];
  float* out = (float*)d_out;

  // workspace (bf16): xb(4M) Wq/Wk/Wv/Wo(1M each) Q,K,Vt,ctx(4M each) = 48 MB
  const size_t big = (size_t)NROWS * DMODEL;   // 4M elems
  const size_t wsz = (size_t)DMODEL * DMODEL;  // 1M elems
  __bf16* xb  = (__bf16*)d_ws;
  __bf16* Wqb = xb + big;
  __bf16* Wkb = Wqb + wsz;
  __bf16* Wvb = Wkb + wsz;
  __bf16* Wob = Wvb + wsz;
  __bf16* Qb  = Wob + wsz;
  __bf16* Kb  = Qb + big;
  __bf16* Vtb = Kb + big;
  __bf16* Cb  = Vtb + big;

  tobf16_kernel<<<(int)(big / 8 / 256), 256, 0, stream>>>(x, xb, (int)(big / 8));
  tobf16_kernel<<<(int)(wsz / 8 / 256), 256, 0, stream>>>(Wq, Wqb, (int)(wsz / 8));
  tobf16_kernel<<<(int)(wsz / 8 / 256), 256, 0, stream>>>(Wk, Wkb, (int)(wsz / 8));
  tobf16_kernel<<<(int)(wsz / 8 / 256), 256, 0, stream>>>(Wv, Wvb, (int)(wsz / 8));
  tobf16_kernel<<<(int)(wsz / 8 / 256), 256, 0, stream>>>(Wo, Wob, (int)(wsz / 8));

  dim3 g1(NROWS / 64, DMODEL / 128);
  qkv_proj_kernel<0><<<g1, 256, 0, stream>>>(xb, Wqb, bq, Qb);
  qkv_proj_kernel<1><<<g1, 256, 0, stream>>>(xb, Wkb, bk, Kb);
  qkv_proj_kernel<2><<<g1, 256, 0, stream>>>(xb, Wvb, bv, Vtb);

  dim3 g2(NTOK / 128, NBATCH * NHEADS);
  attn_kernel<<<g2, 128, 0, stream>>>(Qb, Kb, Vtb, Cb);

  oproj_kernel<<<g1, 256, 0, stream>>>(Cb, Wob, bo, x, out);
  ln_kernel<<<NROWS, 256, 0, stream>>>(out, g, be);
}
